// my_model1_89472758711049
// MI455X (gfx1250) — matrix operand order repaired and checked by reading, not compiled, not run on hardware
//
#include <hip/hip_runtime.h>
#include <cstdint>
#include <cstddef>

#define EPSV 1e-20f

typedef _Float16 v8h  __attribute__((ext_vector_type(8)));
typedef _Float16 v16h __attribute__((ext_vector_type(16)));
typedef float    v8f  __attribute__((ext_vector_type(8)));

__device__ __forceinline__ v16h ld2x8(const _Float16* p0, const _Float16* p1) {
  v8h a = *(const v8h*)p0;
  v8h b = *(const v8h*)p1;
  return __builtin_shufflevector(a, b, 0, 1, 2, 3, 4, 5, 6, 7,
                                 8, 9, 10, 11, 12, 13, 14, 15);
}

__device__ __forceinline__ v16h cat8(v8h a, v8h b) {
  return __builtin_shufflevector(a, b, 0, 1, 2, 3, 4, 5, 6, 7,
                                 8, 9, 10, 11, 12, 13, 14, 15);
}

__device__ __forceinline__ float block_sum(float v) {
  __shared__ float red[256];
  red[threadIdx.x] = v;
  __syncthreads();
  for (int s = 128; s > 0; s >>= 1) {
    if ((int)threadIdx.x < s) red[threadIdx.x] += red[threadIdx.x + s];
    __syncthreads();
  }
  float r = red[0];
  __syncthreads();
  return r;
}

// ---------------------------------------------------------------------------
// Generic batched WMMA GEMM: C[z] = act(A[z] @ B[z] + bias) (row-major).
// A may be split into two sources at column ksplit (concat along K); ksplit
// must be a multiple of 64 (or >= K). Optional relu on A, tanh on output,
// per-row scale, f16 output. Block 256 (8 waves), WG tile 32(M) x 64(N),
// K-chunk 64 (2 chained WMMAs per barrier pair). Branch-free clamped staging.
// ---------------------------------------------------------------------------
__global__ __launch_bounds__(256) void k_gemm(
    const float* __restrict__ A, long long sA, int dA, int lda,
    const float* __restrict__ A2, long long sA2, int lda2, int ksplit, int actA,
    const float* __restrict__ Bm, long long sB, int ldb,
    const float* __restrict__ bias,
    const float* __restrict__ rowscale, long long sR,
    float* __restrict__ C, _Float16* __restrict__ C16, long long sC, int ldc,
    int act, int M, int N, int K)
{
  __shared__ _Float16 As[32][72];   // [m][k]  (72-half stride: 16B aligned)
  __shared__ _Float16 Bs[64][72];   // [n][k]  (transposed: frags contiguous)
  const int tid   = threadIdx.x;
  const int z     = blockIdx.z;
  const int tileM = blockIdx.y * 32;
  const int tileN = blockIdx.x * 64;
  const int lane  = tid & 31;
  const int wave  = tid >> 5;
  const int psub  = wave >> 2;      // 0..1
  const int osub  = wave & 3;       // 0..3
  const int mm    = lane & 15;
  const int hi    = lane >> 4;      // 0 or 1
  const long long zA  = (long long)(z / dA) * sA;
  const long long zA2 = (long long)z * sA2;
  const long long zB  = (long long)z * sB;
  const long long zC  = (long long)z * sC;

  // A staging ids: 32 rows x 64 k, 8 k per thread (two float4 loads)
  const int  ar      = tid >> 3;
  const int  akq     = (tid & 7) * 8;
  const int  arow    = tileM + ar;
  const bool arow_ok = arow < M;
  const int  arow_c  = arow_ok ? arow : (M - 1);
  // B staging ids: 64 n x 64 k, k-major (coalesced global, v8h LDS stores)
  const int  bn     = tid & 63;
  const int  bkg    = (tid >> 6) * 8;     // 0,8,16,24
  const int  bnn    = tileN + bn;
  const bool bn_ok  = bnn < N;
  const int  bnn_c  = bn_ok ? bnn : (N - 1);

  v8f acc = {};
  for (int k0 = 0; k0 < K; k0 += 64) {
    { // ---- stage A (clamped, select-zero; no exec divergence)
      const float* Asrc = A;
      long long base = zA + (long long)arow_c * lda;
      int kk = k0;
      if (k0 >= ksplit) {             // uniform per chunk (ksplit % 64 == 0)
        Asrc = A2; base = zA2 + (long long)arow_c * lda2; kk = k0 - ksplit;
      }
      const bool k_ok = (k0 + akq) < K;   // K is a multiple of 32
      const int  kg   = k_ok ? (kk + akq) : kk;
      float4 v0 = *(const float4*)&Asrc[base + kg];
      float4 v1 = *(const float4*)&Asrc[base + kg + 4];
      if (actA) {
        v0.x = fmaxf(v0.x, 0.f); v0.y = fmaxf(v0.y, 0.f);
        v0.z = fmaxf(v0.z, 0.f); v0.w = fmaxf(v0.w, 0.f);
        v1.x = fmaxf(v1.x, 0.f); v1.y = fmaxf(v1.y, 0.f);
        v1.z = fmaxf(v1.z, 0.f); v1.w = fmaxf(v1.w, 0.f);
      }
      const bool ok = arow_ok && k_ok;
      v8h h;
      h[0] = (_Float16)(ok ? v0.x : 0.f); h[1] = (_Float16)(ok ? v0.y : 0.f);
      h[2] = (_Float16)(ok ? v0.z : 0.f); h[3] = (_Float16)(ok ? v0.w : 0.f);
      h[4] = (_Float16)(ok ? v1.x : 0.f); h[5] = (_Float16)(ok ? v1.y : 0.f);
      h[6] = (_Float16)(ok ? v1.z : 0.f); h[7] = (_Float16)(ok ? v1.w : 0.f);
      *(v8h*)&As[ar][akq] = h;
    }
    { // ---- stage B transposed: thread owns 8+8 k rows for one n column
      const long long bb = zB + bnn_c;
      v8h b0, b1;
#pragma unroll
      for (int j = 0; j < 8; ++j) {
        float v = Bm[bb + (long long)(k0 + bkg + j) * ldb];
        b0[j] = (_Float16)(bn_ok ? v : 0.f);
      }
      const bool p2 = (k0 + 32) < K;
#pragma unroll
      for (int j = 0; j < 8; ++j) {
        int k2 = p2 ? (k0 + 32 + bkg + j) : k0;
        float v = Bm[bb + (long long)k2 * ldb];
        b1[j] = (_Float16)((bn_ok && p2) ? v : 0.f);
      }
      *(v8h*)&Bs[bn][bkg]      = b0;
      *(v8h*)&Bs[bn][32 + bkg] = b1;
    }
    __syncthreads();
    const _Float16* arow_p = &As[psub * 16 + mm][0];
    const _Float16* brow_p = &Bs[osub * 16 + mm][0];
    v16h af0 = ld2x8(arow_p + hi * 8,      arow_p + 16 + hi * 8);
    v16h af1 = ld2x8(arow_p + 32 + hi * 8, arow_p + 48 + hi * 8);
    v16h bf0 = ld2x8(brow_p + hi * 8,      brow_p + 16 + hi * 8);
    v16h bf1 = ld2x8(brow_p + 32 + hi * 8, brow_p + 48 + hi * 8);
    acc = __builtin_amdgcn_wmma_f32_16x16x32_f16(false, af0, false, bf0, (short)0,
                                                 acc, false, false);
    acc = __builtin_amdgcn_wmma_f32_16x16x32_f16(false, af1, false, bf1, (short)0,
                                                 acc, false, false);
    __syncthreads();
  }
  int col = tileN + osub * 16 + mm;
  if (col < N) {
    float bv   = bias ? bias[col] : 0.f;
    int  rbase = tileM + psub * 16 + hi * 8;
#pragma unroll
    for (int v = 0; v < 8; ++v) {
      int row = rbase + v;
      if (row < M) {
        float x = acc[v] + bv;
        if (act == 1) x = tanhf(x);
        if (rowscale) x *= rowscale[(long long)z * sR + row];
        long long ci = zC + (long long)row * ldc + col;
        if (C16) C16[ci] = (_Float16)x;
        else     C[ci]   = x;
      }
    }
  }
}

// ---------------------------------------------------------------------------
// Group bilinear: out[b,p,o] = sum_{k,c,dd} h[p,kc]*t[p,kdd]*W[o,k,c,dd] + bias
// A (rank-64 blocks) built in registers: t-fragments are c-invariant (hoisted),
// h scalars come 8-at-a-time from one ds_load_b128. W fragments are direct
// 32-byte v16h global loads (L2-resident f16 W). h/t tiles staged to LDS via
// CDNA5 async global->LDS copies (ASYNCcnt).
// ---------------------------------------------------------------------------
__global__ __launch_bounds__(256) void k_bili(
    const _Float16* __restrict__ h16, const _Float16* __restrict__ t16,
    const _Float16* __restrict__ W16, const float* __restrict__ bias,
    float* __restrict__ C, int P, int D, int HID)
{
  __shared__ _Float16 hs[32][72];
  __shared__ _Float16 ts[32][72];
  const int tid  = threadIdx.x;
  const int z    = blockIdx.z;
  const int p0   = blockIdx.y * 32;
  const int o0   = blockIdx.x * 64;
  const int lane = tid & 31;
  const int wave = tid >> 5;
  const int psub = wave & 1;
  const int osub = wave >> 1;       // 0..3
  const int mm   = lane & 15;
  const int hi   = lane >> 4;
  const long long hbase = (long long)z * P * D;
  const int orow = o0 + osub * 16 + mm;
  const _Float16* wrow = W16 + (long long)orow * (12 * 64 * 64) + hi * 16;

  const int r  = tid >> 3;          // 0..31
  const int cq = (tid & 7) * 8;     // 0..56
  const unsigned lds_h = (unsigned)(size_t)&hs[r][cq];
  const unsigned lds_t = (unsigned)(size_t)&ts[r][cq];
  const long long grow = hbase + (long long)(p0 + r) * D + cq;

  v8f acc = {};
  for (int k = 0; k < 12; ++k) {
    // ---- async f16 global -> LDS staging (16B per thread per tile)
    {
      const _Float16* gh = h16 + grow + k * 64;
      const _Float16* gt = t16 + grow + k * 64;
      asm volatile("global_load_async_to_lds_b128 %0, %1, off"
                   :: "v"(lds_h), "v"(gh) : "memory");
      asm volatile("global_load_async_to_lds_b128 %0, %1, off"
                   :: "v"(lds_t), "v"(gt) : "memory");
      asm volatile("s_wait_asynccnt 0x0" ::: "memory");
    }
    __syncthreads();
    const _Float16* trow = &ts[psub * 16 + mm][0];
    const _Float16* hrow = &hs[psub * 16 + mm][0];
    // t fragments: invariant across c — load once per k
    v8h t00 = *(const v8h*)(trow + hi * 16);
    v8h t01 = *(const v8h*)(trow + hi * 16 + 8);
    v8h t10 = *(const v8h*)(trow + 32 + hi * 16);
    v8h t11 = *(const v8h*)(trow + 32 + hi * 16 + 8);
    const _Float16* wk = wrow + k * 4096;
    if (k < 11) __builtin_prefetch(wk + 4096, 0, 1);  // next W block -> L2
    for (int j = 0; j < 8; ++j) {
      v8h hvec = *(const v8h*)(hrow + j * 8);         // 8 h scalars, one b128
      const _Float16* wj = wk + j * 8 * 64;
#pragma unroll
      for (int cc = 0; cc < 8; ++cc) {
        _Float16 hv = hvec[cc];                       // constant index
        v8h hv8;
#pragma unroll
        for (int i = 0; i < 8; ++i) hv8[i] = hv;
        v16h af0 = cat8(hv8 * t00, hv8 * t01);
        v16h af1 = cat8(hv8 * t10, hv8 * t11);
        const _Float16* wc = wj + cc * 64;
        v16h bf0 = *(const v16h*)wc;                  // 32B contiguous (dd 0..31)
        v16h bf1 = *(const v16h*)(wc + 32);           // 32B contiguous (dd 32..63)
        acc = __builtin_amdgcn_wmma_f32_16x16x32_f16(false, af0, false, bf0,
                                                     (short)0, acc, false, false);
        acc = __builtin_amdgcn_wmma_f32_16x16x32_f16(false, af1, false, bf1,
                                                     (short)0, acc, false, false);
      }
    }
    __syncthreads();
  }
  const int col   = o0 + osub * 16 + mm;
  const int rbase = p0 + psub * 16 + hi * 8;
  const float bv  = bias[col];
#pragma unroll
  for (int v = 0; v < 8; ++v) {
    long long ci = ((long long)z * P + (rbase + v)) * HID + col;
    C[ci] = acc[v] + bv;
  }
}

// ------------------------------- small kernels ------------------------------
__global__ void k_cvt_half(const float* __restrict__ s, _Float16* __restrict__ d,
                           long long n) {
  long long i = (long long)blockIdx.x * blockDim.x + threadIdx.x;
  long long st = (long long)gridDim.x * blockDim.x;
  for (; i < n; i += st) d[i] = (_Float16)s[i];
}

__global__ void k_exp(const float* __restrict__ s, float* __restrict__ d, long long n) {
  long long i = (long long)blockIdx.x * blockDim.x + threadIdx.x;
  long long st = (long long)gridDim.x * blockDim.x;
  for (; i < n; i += st) d[i] = expf(s[i]);
}

// entity <- log(where(rowsum(emap_row)==0, 1, entity)), in place
__global__ __launch_bounds__(256) void k_entity_log(float* __restrict__ ent,
                                                    const float* __restrict__ emap,
                                                    int Mm, int Dd) {
  int rrow = blockIdx.x;
  float s = 0.f;
  for (int i = threadIdx.x; i < Mm; i += blockDim.x) s += emap[(long long)rrow * Mm + i];
  float tot = block_sum(s);
  bool zero = (tot == 0.f);
  long long base = (long long)rrow * Dd;
  for (int c = threadIdx.x; c < Dd; c += blockDim.x)
    ent[base + c] = zero ? 0.f : logf(ent[base + c]);
}

__global__ __launch_bounds__(256) void k_rownorm(const float* __restrict__ src,
                                                 float* __restrict__ dst, int len) {
  long long base = (long long)blockIdx.x * len;
  float s = 0.f;
  for (int i = threadIdx.x; i < len; i += blockDim.x) s += src[base + i];
  float inv = 1.f / (block_sum(s) + EPSV);
  for (int i = threadIdx.x; i < len; i += blockDim.x) dst[base + i] = src[base + i] * inv;
}

__global__ __launch_bounds__(256) void k_rowsuminv(const float* __restrict__ src,
                                                   float* __restrict__ rs, int len) {
  long long base = (long long)blockIdx.x * len;
  float s = 0.f;
  for (int i = threadIdx.x; i < len; i += blockDim.x) s += src[base + i];
  float tot = block_sum(s);
  if (threadIdx.x == 0) rs[blockIdx.x] = 1.f / (tot + EPSV);
}

// out[b,p,:] = src[b, idx[b,p,sel], :] * mask(b,p)
__global__ __launch_bounds__(256) void k_gather(const float* __restrict__ src,
                                                const int* __restrict__ idx, int sel,
                                                float* __restrict__ out,
                                                int R, int P, int Dd) {
  int bp = blockIdx.x;
  int b  = bp / P;
  int i0 = idx[(long long)bp * 2 + 0];
  int i1 = idx[(long long)bp * 2 + 1];
  float mask = ((i0 + i1) != 0) ? 1.f : 0.f;
  int ii = sel ? i1 : i0;
  long long sbase = ((long long)b * R + ii) * Dd;
  long long obase = (long long)bp * Dd;
  for (int c = threadIdx.x; c < Dd; c += blockDim.x) out[obase + c] = src[sbase + c] * mask;
}

// ca[b,p,l] = mask * (sum_h ea[b,h,i0,l]*ea[b,h,i1,l]) / (sum_l ... + EPS)
__global__ __launch_bounds__(256) void k_ca(const float* __restrict__ ea,
                                            const int* __restrict__ idx,
                                            float* __restrict__ ca,
                                            int R, int P, int Hh, int Ll) {
  int bp = blockIdx.x;
  int b  = bp / P;
  int i0 = idx[(long long)bp * 2 + 0];
  int i1 = idx[(long long)bp * 2 + 1];
  float mask = ((i0 + i1) != 0) ? 1.f : 0.f;
  float vals[4];
  float s = 0.f;
  int cnt = 0;
  for (int l = threadIdx.x; l < Ll; l += blockDim.x) {
    float a = 0.f;
    for (int h = 0; h < Hh; ++h) {
      long long o = (long long)(b * Hh + h) * R;
      a += ea[(o + i0) * Ll + l] * ea[(o + i1) * Ll + l];
    }
    vals[cnt++] = a;
    s += a;
  }
  float scale = mask / (block_sum(s) + EPSV);
  cnt = 0;
  for (int l = threadIdx.x; l < Ll; l += blockDim.x)
    ca[(long long)bp * Ll + l] = vals[cnt++] * scale;
}

// ---------------------------------------------------------------------------
extern "C" void kernel_launch(void* const* d_in, const int* in_sizes, int n_in,
                              void* d_out, int out_size, void* d_ws, size_t ws_size,
                              hipStream_t stream) {
  (void)in_sizes; (void)n_in; (void)out_size; (void)ws_size;
  const float* context   = (const float*)d_in[0];
  const float* attention = (const float*)d_in[1];
  const float* mmap      = (const float*)d_in[2];
  const float* emap      = (const float*)d_in[3];
  const float* mhtmap    = (const float*)d_in[4];
  const float* Wh        = (const float*)d_in[5];
  const float* bh        = (const float*)d_in[6];
  const float* Wbili     = (const float*)d_in[7];
  const float* bbili     = (const float*)d_in[8];
  const float* Wclas     = (const float*)d_in[9];
  const float* bclas     = (const float*)d_in[10];
  const int*   hts       = (const int*)d_in[11];
  const int*   mht       = (const int*)d_in[12];
  float* out = (float*)d_out;

  const int B = 4, L = 512, D = 768, H = 12, Mm = 96, E = 48;
  const int PE = 1024, PM = 1024, HID = 768, REL = 97;

  char* w = (char*)d_ws;
  size_t off = 0;
  auto alloc_f = [&](size_t n) {
    float* p = (float*)(w + off);
    off = (off + n * 4 + 255) & ~(size_t)255;
    return p;
  };
  auto alloc_h = [&](size_t n) {
    _Float16* p = (_Float16*)(w + off);
    off = (off + n * 2 + 255) & ~(size_t)255;
    return p;
  };

  float* mention = alloc_f((size_t)B * Mm * D);
  float* expm    = alloc_f((size_t)B * Mm * D);
  float* entity  = alloc_f((size_t)B * E * D);
  float* emap_e  = alloc_f((size_t)B * E * L);
  float* emap_m  = alloc_f((size_t)B * Mm * L);
  float* ea_e    = alloc_f((size_t)B * H * E * L);
  float* ea_m    = alloc_f((size_t)B * H * Mm * L);
  float* ca_e    = alloc_f((size_t)B * PE * L);
  float* ca_m    = alloc_f((size_t)B * PM * L);
  float* e_ctx   = alloc_f((size_t)B * PE * D);
  float* m_ctx   = alloc_f((size_t)B * PM * D);
  float* gath    = alloc_f((size_t)B * PE * D);
  float* e_logit = alloc_f((size_t)B * PE * HID);
  float* m_logit = alloc_f((size_t)B * PM * HID);
  float* m_pool  = alloc_f((size_t)B * PE * HID);
  float* mrs     = alloc_f((size_t)B * PE);
  _Float16* eh16 = alloc_h((size_t)B * PE * D);
  _Float16* et16 = alloc_h((size_t)B * PE * D);
  _Float16* mh16 = alloc_h((size_t)B * PM * D);
  _Float16* mt16 = alloc_h((size_t)B * PM * D);
  _Float16* W16  = alloc_h((size_t)HID * 12 * 64 * 64);

  auto gemm = [&](const float* A, long long sA, int dA, int lda,
                  const float* A2, long long sA2, int lda2, int ksplit, int actA,
                  const float* Bp, long long sB, int ldb,
                  const float* bias, const float* rowscale, long long sR,
                  float* C, _Float16* C16, long long sC, int ldc,
                  int act, int M, int N, int K, int Z) {
    dim3 g((N + 63) / 64, (M + 31) / 32, Z);
    k_gemm<<<g, 256, 0, stream>>>(A, sA, dA, lda, A2, sA2, lda2, ksplit, actA,
                                  Bp, sB, ldb, bias, rowscale, sR,
                                  C, C16, sC, ldc, act, M, N, K);
  };
  const int BIG = 1 << 30;

  // Wbili -> f16 (L2-resident B operand for k_bili)
  k_cvt_half<<<2048, 256, 0, stream>>>(Wbili, W16, (long long)HID * 12 * 64 * 64);

  // mention = mention_map @ context
  gemm(mmap, (long long)Mm * L, 1, L, nullptr, 0, 0, BIG, 0,
       context, (long long)L * D, D, nullptr, nullptr, 0,
       mention, nullptr, (long long)Mm * D, D, 0, Mm, D, L, B);
  k_exp<<<1024, 256, 0, stream>>>(mention, expm, (long long)B * Mm * D);

  // entity = entity_map @ exp(mention); then masked log (in place)
  gemm(emap, (long long)E * Mm, 1, Mm, nullptr, 0, 0, BIG, 0,
       expm, (long long)Mm * D, D, nullptr, nullptr, 0,
       entity, nullptr, (long long)E * D, D, 0, E, D, Mm, B);
  k_entity_log<<<B * E, 256, 0, stream>>>(entity, emap, Mm, D);

  // emap_e = rownorm(entity_map @ mention_map); emap_m = rownorm(mention_map)
  gemm(emap, (long long)E * Mm, 1, Mm, nullptr, 0, 0, BIG, 0,
       mmap, (long long)Mm * L, L, nullptr, nullptr, 0,
       emap_e, nullptr, (long long)E * L, L, 0, E, L, Mm, B);
  k_rownorm<<<B * E, 256, 0, stream>>>(emap_e, emap_e, L);
  k_rownorm<<<B * Mm, 256, 0, stream>>>(mmap, emap_m, L);

  // ea = einsum('bel,bhlm->bhem') as batched GEMM over z = b*H+h
  gemm(emap_e, (long long)E * L, H, L, nullptr, 0, 0, BIG, 0,
       attention, (long long)L * L, L, nullptr, nullptr, 0,
       ea_e, nullptr, (long long)E * L, L, 0, E, L, L, B * H);
  gemm(emap_m, (long long)Mm * L, H, L, nullptr, 0, 0, BIG, 0,
       attention, (long long)L * L, L, nullptr, nullptr, 0,
       ea_m, nullptr, (long long)Mm * L, L, 0, Mm, L, L, B * H);

  // ca = normalize(sum_h ea[h]*ea[t]) * mask
  k_ca<<<B * PE, 256, 0, stream>>>(ea_e, hts, ca_e, E, PE, H, L);
  k_ca<<<B * PM, 256, 0, stream>>>(ea_m, mht, ca_m, Mm, PM, H, L);

  // ctx = ca @ context
  gemm(ca_e, (long long)PE * L, 1, L, nullptr, 0, 0, BIG, 0,
       context, (long long)L * D, D, nullptr, nullptr, 0,
       e_ctx, nullptr, (long long)PE * D, D, 0, PE, D, L, B);
  gemm(ca_m, (long long)PM * L, 1, L, nullptr, 0, 0, BIG, 0,
       context, (long long)L * D, D, nullptr, nullptr, 0,
       m_ctx, nullptr, (long long)PM * D, D, 0, PM, D, L, B);

  // entity-pair reps: tanh(concat(gather, ctx) @ Wh + bh) -> f16
  k_gather<<<B * PE, 256, 0, stream>>>(entity, hts, 0, gath, E, PE, D);
  gemm(gath, (long long)PE * D, 1, D, e_ctx, (long long)PE * D, D, D, 0,
       Wh, 0, D, bh, nullptr, 0,
       nullptr, eh16, (long long)PE * D, D, 1, PE, D, 2 * D, B);
  k_gather<<<B * PE, 256, 0, stream>>>(entity, hts, 1, gath, E, PE, D);
  gemm(gath, (long long)PE * D, 1, D, e_ctx, (long long)PE * D, D, D, 0,
       Wh, 0, D, bh, nullptr, 0,
       nullptr, et16, (long long)PE * D, D, 1, PE, D, 2 * D, B);
  k_bili<<<dim3(HID / 64, PE / 32, B), 256, 0, stream>>>(eh16, et16, W16, bbili,
                                                         e_logit, PE, D, HID);

  // mention-pair reps
  k_gather<<<B * PM, 256, 0, stream>>>(mention, mht, 0, gath, Mm, PM, D);
  gemm(gath, (long long)PM * D, 1, D, m_ctx, (long long)PM * D, D, D, 0,
       Wh, 0, D, bh, nullptr, 0,
       nullptr, mh16, (long long)PM * D, D, 1, PM, D, 2 * D, B);
  k_gather<<<B * PM, 256, 0, stream>>>(mention, mht, 1, gath, Mm, PM, D);
  gemm(gath, (long long)PM * D, 1, D, m_ctx, (long long)PM * D, D, D, 0,
       Wh, 0, D, bh, nullptr, 0,
       nullptr, mt16, (long long)PM * D, D, 1, PM, D, 2 * D, B);
  k_bili<<<dim3(HID / 64, PM / 32, B), 256, 0, stream>>>(mh16, mt16, W16, bbili,
                                                         m_logit, PM, D, HID);

  // m_pool = (mention_ht_map @ m_logit) / (rowsum + EPS)
  k_rowsuminv<<<B * PE, 256, 0, stream>>>(mhtmap, mrs, PM);
  gemm(mhtmap, (long long)PE * PM, 1, PM, nullptr, 0, 0, BIG, 0,
       m_logit, (long long)PM * HID, HID, nullptr, mrs, (long long)PE,
       m_pool, nullptr, (long long)PE * HID, HID, 0, PE, HID, PM, B);

  // out = relu(concat(e_logit, m_pool)) @ Wclas + bclas
  gemm(e_logit, (long long)PE * HID, 1, HID, m_pool, (long long)PE * HID, HID, HID, 1,
       Wclas, 0, REL, bclas, nullptr, 0,
       out, nullptr, (long long)PE * REL, REL, 0, PE, REL, 2 * HID, B);
}
